// MoeLayer_32332513804974
// MI455X (gfx1250) — matrix operand order repaired and checked by reading, not compiled, not run on hardware
//
#include <hip/hip_runtime.h>
#include <hip/hip_bf16.h>

// ---------------- problem constants ----------------
#define HDIM   1024
#define DFF    2048
#define NE     8
#define NTOK   2048      // B*S
#define MAXSLOT 2048
#define OUT_N  (NTOK * HDIM)

typedef __bf16 bf16_t;
typedef __bf16 v16bf __attribute__((ext_vector_type(16)));
typedef __bf16 v8bf  __attribute__((ext_vector_type(8)));
typedef __bf16 v4bf  __attribute__((ext_vector_type(4)));
typedef float  v8f   __attribute__((ext_vector_type(8)));
typedef float  v4f   __attribute__((ext_vector_type(4)));

// ---------------- WMMA helpers ----------------
__device__ __forceinline__ v8f wmma_bf16(v16bf a, v16bf b, v8f c) {
  // D = A(16x32 bf16) * B(32x16 bf16) + C(16x16 f32)
  return __builtin_amdgcn_wmma_f32_16x16x32_bf16(
      /*neg_a=*/false, a, /*neg_b=*/false, b,
      /*c_mod=*/(short)0, c, /*reuse_a=*/false, /*reuse_b=*/false);
}

// A fragment (16x32 bf16) from a bf16 row (this lane's M-row), 2x 16B loads.
// lane<16: elems0..15 = K k0..k0+15; lane>=16: K k0+16..k0+31 (matches B).   half8 = (lane>=16)?8:0
__device__ __forceinline__ v16bf a_frag_bf16(const bf16_t* rowptr, int k0, int half8) {
  const v8bf* p0 = (const v8bf*)(rowptr + k0 + 2 * half8);
  const v8bf* p1 = (const v8bf*)(rowptr + k0 + 2 * half8 + 8);
  v8bf lo = *p0, hi = *p1;
  v16bf a;
#pragma unroll
  for (int j = 0; j < 8; ++j) { a[j] = lo[j]; a[8 + j] = hi[j]; }
  return a;
}

// B fragment (32x16 bf16) from an N-major (pre-transposed) bf16 weight row.
// rowptr = &Wt[n][0] for this lane's column n; kb = k0 + ((lane>=16)?16:0).
// lane<16: K = k0..k0+15, lane>=16: K = k0+16..k0+31 -> 16 contiguous bf16.
__device__ __forceinline__ v16bf b_frag_row(const bf16_t* rowptr, int kb) {
  const v8bf* q = (const v8bf*)(rowptr + kb);
  v8bf lo = q[0], hi = q[1];
  v16bf b;
#pragma unroll
  for (int j = 0; j < 8; ++j) { b[j] = lo[j]; b[8 + j] = hi[j]; }
  return b;
}

// ---------------- utility kernels ----------------
__global__ void moe_zero(float* __restrict__ out, int n) {
  int i = blockIdx.x * blockDim.x + threadIdx.x;
  if (i < n) out[i] = 0.0f;
}

// x (fp32) -> xb (bf16), 4 elements/thread
__global__ void moe_cvt_x(const float* __restrict__ x, bf16_t* __restrict__ xb) {
  int i = (blockIdx.x * blockDim.x + threadIdx.x) * 4;
  v4f f = *(const v4f*)(x + i);
  v4bf o;
#pragma unroll
  for (int j = 0; j < 4; ++j) o[j] = (bf16_t)f[j];
  *(v4bf*)(xb + i) = o;
}

// W (fp32, [K][N] row-major, per expert) -> Wt (bf16, [N][K] row-major).
// 32x32 LDS tile transpose; grid = (N/32, K/32, NE).
__global__ void moe_transpose_cvt(const float* __restrict__ src, bf16_t* __restrict__ dst,
                                  int K, int N) {
  __shared__ float tile[32][33];
  size_t eoff = (size_t)blockIdx.z * K * N;
  const float* s = src + eoff;
  bf16_t* d = dst + eoff;
  int n0 = blockIdx.x * 32, k0 = blockIdx.y * 32;
  int tx = threadIdx.x & 31, ty = threadIdx.x >> 5;   // 32 x 8
#pragma unroll
  for (int r = 0; r < 32; r += 8)
    tile[ty + r][tx] = s[(size_t)(k0 + ty + r) * N + n0 + tx];
  __syncthreads();
#pragma unroll
  for (int r = 0; r < 32; r += 8)
    d[(size_t)(n0 + ty + r) * K + k0 + tx] = (bf16_t)tile[tx][ty + r];
}

// One wave32 per token: logits = x . gate_w + gate_b, top-2, softmax.
__global__ void moe_gate(const float* __restrict__ x, const float* __restrict__ gw,
                         const float* __restrict__ gb,
                         int* __restrict__ topi, float* __restrict__ topw) {
  int wave = threadIdx.x >> 5, lane = threadIdx.x & 31;
  int tok  = blockIdx.x * 8 + wave;
  const float* xr = x + (size_t)tok * HDIM;
  float acc[NE];
#pragma unroll
  for (int e = 0; e < NE; ++e) acc[e] = 0.0f;
  for (int h = lane; h < HDIM; h += 32) {
    float xv = xr[h];
    const v4f* g4 = (const v4f*)(gw + h * NE);
    v4f g0 = g4[0], g1 = g4[1];
#pragma unroll
    for (int j = 0; j < 4; ++j) { acc[j] += xv * g0[j]; acc[4 + j] += xv * g1[j]; }
  }
#pragma unroll
  for (int e = 0; e < NE; ++e)
    for (int off = 16; off; off >>= 1) acc[e] += __shfl_xor(acc[e], off, 32);
  if (lane == 0) {
    float v[NE];
#pragma unroll
    for (int e = 0; e < NE; ++e) v[e] = acc[e] + gb[e];
    int e0 = 0; float v0 = v[0];
#pragma unroll
    for (int e = 1; e < NE; ++e) if (v[e] > v0) { v0 = v[e]; e0 = e; }
    int e1 = -1; float v1 = -3.4e38f;
#pragma unroll
    for (int e = 0; e < NE; ++e) if (e != e0 && v[e] > v1) { v1 = v[e]; e1 = e; }
    float ev = __expf(v1 - v0);         // v0 >= v1
    float p0 = 1.0f / (1.0f + ev);
    float p1 = ev * p0;
    topi[2 * tok] = e0; topi[2 * tok + 1] = e1;
    topw[2 * tok] = p0; topw[2 * tok + 1] = p1;
  }
}

// Deterministic routing: single block, wave w owns expert w; token-order compaction.
__global__ void moe_route(const int* __restrict__ topi, const float* __restrict__ topw,
                          int* __restrict__ tokidx, float* __restrict__ tokw,
                          int* __restrict__ cnt, int* __restrict__ boff) {
  __shared__ int scnt[NE];
  int wave = threadIdx.x >> 5, lane = threadIdx.x & 31;
  if (wave < NE) {
    int e = wave, base = 0;
    for (int t0 = 0; t0 < NTOK; t0 += 32) {
      int t = t0 + lane;
      int a = topi[2 * t], b = topi[2 * t + 1];
      bool hit = (a == e) || (b == e);
      float w = (a == e) ? topw[2 * t] : ((b == e) ? topw[2 * t + 1] : 0.0f);
      unsigned m = (unsigned)__ballot(hit);
      if (hit) {
        int pos = base + __popc(m & ((1u << lane) - 1u));
        tokidx[e * MAXSLOT + pos] = t;
        tokw[e * MAXSLOT + pos]   = w;
      }
      base += __popc(m);
    }
    if (lane == 0) { scnt[e] = base; cnt[e] = base; }
  }
  __syncthreads();
  if (threadIdx.x == 0) {
    int off = 0;
    for (int e = 0; e < NE; ++e) { boff[e] = off; off += (scnt[e] + 15) & ~15; }
  }
}

// Stage 1: Z[16,2048] = (X*W1+b1) .* (X*W2+b2)   (bf16 output to workspace)
__global__ __launch_bounds__(256)
void moe_ffn1(const bf16_t* __restrict__ xb,
              const bf16_t* __restrict__ w1t, const float* __restrict__ b1,
              const bf16_t* __restrict__ w2t, const float* __restrict__ b2,
              const int* __restrict__ tokidx, const int* __restrict__ cnt,
              const int* __restrict__ boff, bf16_t* __restrict__ Z) {
  int e = blockIdx.x >> 7, tile = blockIdx.x & 127;
  int ce = cnt[e];
  if (tile * 16 >= ce) return;                     // uniform over block
  int wave = threadIdx.x >> 5, lane = threadIdx.x & 31;
  int half8   = (lane >> 4) << 3;                  // 0 or 8
  int klane16 = (lane >> 4) << 4;                  // 0 or 16
  int s = tile * 16 + (lane & 15);
  if (s > ce - 1) s = ce - 1;                      // pad partial tile (weight 0 later)
  int tok = tokidx[e * MAXSLOT + s];
  const bf16_t* xrow = xb + (size_t)tok * HDIM;
  const bf16_t* W1 = w1t + (size_t)e * HDIM * DFF; // [DFF][HDIM] N-major
  const bf16_t* W2 = w2t + (size_t)e * HDIM * DFF;
  int zrow0 = boff[e] + tile * 16;

  for (int t = 0; t < 16; ++t) {                   // wave covers 256 contiguous cols
    int n0 = wave * 256 + t * 16;
    const bf16_t* r1 = W1 + (size_t)(n0 + (lane & 15)) * HDIM;
    const bf16_t* r2 = W2 + (size_t)(n0 + (lane & 15)) * HDIM;
    __builtin_prefetch(r1 + klane16, 0, 1);        // global_prefetch_b8
    __builtin_prefetch(r2 + klane16, 0, 1);
    v8f c1 = {}, c2 = {};
    for (int k0 = 0; k0 < HDIM; k0 += 32) {
      v16bf a  = a_frag_bf16(xrow, k0, half8);
      v16bf bA = b_frag_row(r1, k0 + klane16);
      v16bf bB = b_frag_row(r2, k0 + klane16);
      c1 = wmma_bf16(a, bA, c1);
      c2 = wmma_bf16(a, bB, c2);
    }
    int n = n0 + (lane & 15);
    float bv1 = b1[e * DFF + n], bv2 = b2[e * DFF + n];
#pragma unroll
    for (int i = 0; i < 8; ++i) {                  // D layout: m = i + half8, col = n
      float z = (c1[i] + bv1) * (c2[i] + bv2);
      Z[(size_t)(zrow0 + i + half8) * DFF + n] = (bf16_t)z;
    }
  }
}

// Stage 2+3 fused: S = silu(Z*Wg+bg) per 256-col block (LDS), Y += S*W3, scatter-add.
__global__ __launch_bounds__(256)
void moe_ffn2(const bf16_t* __restrict__ Z,
              const bf16_t* __restrict__ wgt, const float* __restrict__ bg,
              const bf16_t* __restrict__ w3t, const float* __restrict__ b3,
              const int* __restrict__ tokidx, const float* __restrict__ tokw,
              const int* __restrict__ cnt, const int* __restrict__ boff,
              float* __restrict__ out) {
  __shared__ __align__(16) bf16_t Sblk[16][264];   // 264*2B row stride = 16B multiple
  int e = blockIdx.x >> 7, tile = blockIdx.x & 127;
  int ce = cnt[e];
  if (tile * 16 >= ce) return;                     // uniform over block
  int wave = threadIdx.x >> 5, lane = threadIdx.x & 31;
  int half8   = (lane >> 4) << 3;
  int klane16 = (lane >> 4) << 4;
  const bf16_t* zrow = Z + (size_t)(boff[e] + tile * 16 + (lane & 15)) * DFF;
  const bf16_t* WG = wgt + (size_t)e * DFF * DFF;  // [g][f] N-major
  const bf16_t* W3 = w3t + (size_t)e * DFF * HDIM; // [h][g] N-major

  const bf16_t* r3[8];
#pragma unroll
  for (int tt = 0; tt < 8; ++tt)
    r3[tt] = W3 + (size_t)(wave * 128 + tt * 16 + (lane & 15)) * DFF;

  v8f y[8];
#pragma unroll
  for (int t = 0; t < 8; ++t) y[t] = (v8f){};

  for (int g = 0; g < 8; ++g) {
    int gbase = g * 256;
    // ---- stage 2: S block (silu) into LDS ----
#pragma unroll
    for (int tt = 0; tt < 2; ++tt) {
      int nl = wave * 32 + tt * 16;                // local col in S block
      int ng = gbase + nl;                         // global DFF col
      const bf16_t* rg = WG + (size_t)(ng + (lane & 15)) * DFF;
      __builtin_prefetch(rg + klane16, 0, 1);      // global_prefetch_b8
      v8f c = {};
      for (int k0 = 0; k0 < DFF; k0 += 32) {
        v16bf a = a_frag_bf16(zrow, k0, half8);
        v16bf b = b_frag_row(rg, k0 + klane16);
        c = wmma_bf16(a, b, c);
      }
      int nle = nl + (lane & 15);
      float bgv = bg[e * DFF + ng + (lane & 15)];
#pragma unroll
      for (int i = 0; i < 8; ++i) {
        float v  = c[i] + bgv;
        float sv = v / (1.0f + __expf(-v));        // silu
        Sblk[i + half8][nle] = (bf16_t)sv;
      }
    }
    __syncthreads();
    // ---- stage 3: Y += S_block * W3[gbase:gbase+256, :] ----
    const bf16_t* srow = &Sblk[lane & 15][0];
    for (int kk = 0; kk < 256; kk += 32) {
      v16bf a = a_frag_bf16(srow, kk, half8);      // ds_load_b128 x2
#pragma unroll
      for (int tt = 0; tt < 8; ++tt) {
        v16bf b = b_frag_row(r3[tt], gbase + kk + klane16);
        y[tt] = wmma_bf16(a, b, y[tt]);
      }
    }
    __syncthreads();
  }
  // ---- epilogue: weighted scatter-add (exactly 2 adds per out element) ----
#pragma unroll
  for (int tt = 0; tt < 8; ++tt) {
    int n = wave * 128 + tt * 16 + (lane & 15);
    float b3v = b3[e * HDIM + n];
#pragma unroll
    for (int i = 0; i < 8; ++i) {
      int s = tile * 16 + i + half8;
      if (s < ce) {
        int tok = tokidx[e * MAXSLOT + s];
        float w = tokw[e * MAXSLOT + s];
        unsafeAtomicAdd(out + (size_t)tok * HDIM + n, w * (y[tt][i] + b3v));
      }
    }
  }
}

// ---------------- launch ----------------
extern "C" void kernel_launch(void* const* d_in, const int* in_sizes, int n_in,
                              void* d_out, int out_size, void* d_ws, size_t ws_size,
                              hipStream_t stream) {
  (void)in_sizes; (void)n_in; (void)out_size; (void)ws_size;
  const float* x   = (const float*)d_in[0];
  const float* gw  = (const float*)d_in[1];
  const float* gb  = (const float*)d_in[2];
  const float* w1  = (const float*)d_in[3];
  const float* b1  = (const float*)d_in[4];
  const float* w2  = (const float*)d_in[5];
  const float* b2  = (const float*)d_in[6];
  const float* wg  = (const float*)d_in[7];
  const float* bg  = (const float*)d_in[8];
  const float* w3  = (const float*)d_in[9];
  const float* b3  = (const float*)d_in[10];
  float* out = (float*)d_out;

  // workspace layout (all 256B aligned except boff)
  char* ws = (char*)d_ws;
  int*    topi   = (int*)(ws + 0);              // 16 KB
  float*  topw   = (float*)(ws + 16384);        // 16 KB
  int*    tokidx = (int*)(ws + 32768);          // 64 KB
  float*  tokw   = (float*)(ws + 98304);        // 64 KB
  int*    cnt    = (int*)(ws + 163840);
  int*    boff   = (int*)(ws + 163872);
  bf16_t* xb     = (bf16_t*)(ws + 164096);      // 4 MB      (2048x1024 bf16)
  bf16_t* Zbuf   = (bf16_t*)(ws + 4358400ull);  // 16.5 MB   (4224x2048 bf16)
  bf16_t* w1t    = (bf16_t*)(ws + 21659904ull); // 32 MB     (8x[2048][1024] bf16)
  bf16_t* w2t    = (bf16_t*)(ws + 55214336ull); // 32 MB
  bf16_t* wgt    = (bf16_t*)(ws + 88768768ull); // 64 MB     (8x[2048][2048] bf16)
  bf16_t* w3t    = (bf16_t*)(ws + 155877632ull);// 32 MB     (8x[1024][2048] bf16)

  moe_zero <<<dim3((OUT_N + 255) / 256), dim3(256), 0, stream>>>(out, OUT_N);
  moe_cvt_x<<<dim3(OUT_N / 4 / 256), dim3(256), 0, stream>>>(x, xb);
  moe_transpose_cvt<<<dim3(DFF / 32, HDIM / 32, NE), dim3(256), 0, stream>>>(w1, w1t, HDIM, DFF);
  moe_transpose_cvt<<<dim3(DFF / 32, HDIM / 32, NE), dim3(256), 0, stream>>>(w2, w2t, HDIM, DFF);
  moe_transpose_cvt<<<dim3(DFF / 32, DFF / 32, NE), dim3(256), 0, stream>>>(wg, wgt, DFF, DFF);
  moe_transpose_cvt<<<dim3(HDIM / 32, DFF / 32, NE), dim3(256), 0, stream>>>(w3, w3t, DFF, HDIM);
  moe_gate <<<dim3(NTOK / 8), dim3(256), 0, stream>>>(x, gw, gb, topi, topw);
  moe_route<<<dim3(1), dim3(256), 0, stream>>>(topi, topw, tokidx, tokw, cnt, boff);
  moe_ffn1 <<<dim3(NE * 128), dim3(256), 0, stream>>>(xb, w1t, b1, w2t, b2,
                                                      tokidx, cnt, boff, Zbuf);
  moe_ffn2 <<<dim3(NE * 128), dim3(256), 0, stream>>>(Zbuf, wgt, bg, w3t, b3,
                                                      tokidx, tokw, cnt, boff, out);
}
